// HashGridMaterial_29884382445934
// MI455X (gfx1250) — matrix operand order repaired and checked by reading, not compiled, not run on hardware
//
#include <hip/hip_runtime.h>
#include <hip/hip_bf16.h>
#include <math.h>

typedef __attribute__((ext_vector_type(16))) _Float16 v16h;
typedef __attribute__((ext_vector_type(8)))  _Float16 v8h;
typedef __attribute__((ext_vector_type(8)))  float    v8f;

#define LOG2_T        18
#define TABLE_MASK    ((1u << LOG2_T) - 1u)
#define PTS_PER_BLOCK 128
#define FEAT_STRIDE   72   // 64 feats + 8 halves pad (144B rows -> spread LDS banks)

__device__ __forceinline__ v8f wmma16x16x32(v16h a, v16h b, v8f c) {
    // (neg_a, A, neg_b, B, c_mod, C, reuse_a, reuse_b)
    return __builtin_amdgcn_wmma_f32_16x16x32_f16(false, a, false, b, (short)0, c,
                                                  false, false);
}

__global__ __launch_bounds__(256)
void hashgrid_mlp_fused(const float* __restrict__ ipos,
                        const float* __restrict__ tables,
                        const float* __restrict__ W1, const float* __restrict__ b1,
                        const float* __restrict__ W2, const float* __restrict__ b2,
                        const float* __restrict__ W3, const float* __restrict__ b3,
                        float* __restrict__ out, int nPoints)
{
    __shared__ __align__(32) _Float16 sW1[64 * 64];
    __shared__ __align__(32) _Float16 sW2[64 * 64];
    __shared__ __align__(32) _Float16 sW3[16 * 64];            // rows 8..15 zero
    __shared__ __align__(32) _Float16 sA[PTS_PER_BLOCK * FEAT_STRIDE];
    __shared__ __align__(32) _Float16 sB[PTS_PER_BLOCK * FEAT_STRIDE];

    const int t = threadIdx.x;

    // ---------- stage weights f32 -> f16 into LDS ----------
    #pragma unroll
    for (int i = t; i < 64 * 64; i += 256) {
        sW1[i] = (_Float16)W1[i];
        sW2[i] = (_Float16)W2[i];
    }
    for (int i = t; i < 16 * 64; i += 256)
        sW3[i] = (i < 8 * 64) ? (_Float16)W3[i] : (_Float16)0.0f;

    // ---------- hash-grid encode: 2 threads per point, 8 levels each ----------
    // res[lvl] = floor(16 * 2^(lvl/3))  (growth = 32^(1/15) = 2^(1/3))
    static constexpr int kRes[16] = {16, 20, 25, 32, 40, 50, 64, 80,
                                     101, 128, 161, 203, 256, 322, 406, 512};

    const int  p  = t >> 1;        // local point 0..127
    const int  lh = t & 1;         // level half: levels [8*lh, 8*lh+8)
    const long gp = (long)blockIdx.x * PTS_PER_BLOCK + p;

    float px = 0.f, py = 0.f, pz = 0.f;
    if (gp < nPoints) {
        px = ipos[gp * 3 + 0];
        py = ipos[gp * 3 + 1];
        pz = ipos[gp * 3 + 2];
    }

    #pragma unroll
    for (int l = 0; l < 8; ++l) {
        const int   lvl = lh * 8 + l;
        const float res = (float)kRes[lvl];
        const float xlx = px * res, xly = py * res, xlz = pz * res;
        const float fx = floorf(xlx), fy = floorf(xly), fz = floorf(xlz);
        const float xfx = xlx - fx, xfy = xly - fy, xfz = xlz - fz;
        const unsigned xix = (unsigned)(int)fx;
        const unsigned xiy = (unsigned)(int)fy;
        const unsigned xiz = (unsigned)(int)fz;
        const float4* tbl = (const float4*)tables + ((size_t)lvl << LOG2_T);

        float a0 = 0.f, a1 = 0.f, a2 = 0.f, a3 = 0.f;
        #pragma unroll
        for (int c = 0; c < 8; ++c) {
            const unsigned cx = xix + ((c >> 2) & 1);
            const unsigned cy = xiy + ((c >> 1) & 1);
            const unsigned cz = xiz + (c & 1);
            const float w = ((c & 4) ? xfx : 1.f - xfx) *
                            ((c & 2) ? xfy : 1.f - xfy) *
                            ((c & 1) ? xfz : 1.f - xfz);
            const unsigned h =
                (cx ^ (cy * 2654435761u) ^ (cz * 805459861u)) & TABLE_MASK;
            const float4 v = tbl[h];                 // global_load_b128 gather (L2-resident)
            a0 += w * v.x; a1 += w * v.y; a2 += w * v.z; a3 += w * v.w;
        }
        _Float16* dst = &sA[p * FEAT_STRIDE + lvl * 4];
        dst[0] = (_Float16)a0; dst[1] = (_Float16)a1;
        dst[2] = (_Float16)a2; dst[3] = (_Float16)a3;
    }

    __syncthreads();

    // ---------- fused 3-layer MLP via WMMA (each wave owns 16 points) ----------
    const int lane = t & 31;
    const int wv   = t >> 5;        // wave 0..7
    const int nl   = lane & 15;     // N (or M-row select for A)
    const int hs   = lane >> 4;     // lane half-group
    const int row0 = wv * 16;       // this wave's point rows in the LDS tiles

    // A-fragment: 16-bit A 16x32 layout -> lane holds K = 16*hs..+16 (contiguous, matches B)
    auto loadA = [&](const _Float16* buf, int kc) -> v16h {
        const _Float16* pb = buf + (row0 + nl) * FEAT_STRIDE + kc * 32 + hs * 16;
        v8h lo = *(const v8h*)(pb);
        v8h hi = *(const v8h*)(pb + 8);
        return __builtin_shufflevector(lo, hi, 0, 1, 2, 3, 4, 5, 6, 7,
                                       8, 9, 10, 11, 12, 13, 14, 15);
    };
    // B-fragment: 16-bit B 32x16 layout -> lane = col N, 16 contiguous K halves
    auto loadB = [&](const _Float16* w, int nt, int kc) -> v16h {
        return *(const v16h*)(w + (nt * 16 + nl) * 64 + kc * 32 + hs * 16);
    };

    // ----- layer 1: sA(features) -> sB = relu(X @ W1^T + b1) -----
    {
        const v16h a0 = loadA(sA, 0);
        const v16h a1 = loadA(sA, 1);
        #pragma unroll
        for (int nt = 0; nt < 4; ++nt) {
            const float bv = b1[nt * 16 + nl];
            v8f c;
            #pragma unroll
            for (int i = 0; i < 8; ++i) c[i] = bv;
            c = wmma16x16x32(a0, loadB(sW1, nt, 0), c);
            c = wmma16x16x32(a1, loadB(sW1, nt, 1), c);
            #pragma unroll
            for (int i = 0; i < 8; ++i) {
                const float r = fmaxf(c[i], 0.f);
                sB[(row0 + hs * 8 + i) * FEAT_STRIDE + nt * 16 + nl] = (_Float16)r;
            }
        }
    }

    // ----- layer 2: sB -> sA = relu(H1 @ W2^T + b2)  (rows are wave-private) -----
    {
        const v16h a0 = loadA(sB, 0);
        const v16h a1 = loadA(sB, 1);
        #pragma unroll
        for (int nt = 0; nt < 4; ++nt) {
            const float bv = b2[nt * 16 + nl];
            v8f c;
            #pragma unroll
            for (int i = 0; i < 8; ++i) c[i] = bv;
            c = wmma16x16x32(a0, loadB(sW2, nt, 0), c);
            c = wmma16x16x32(a1, loadB(sW2, nt, 1), c);
            #pragma unroll
            for (int i = 0; i < 8; ++i) {
                const float r = fmaxf(c[i], 0.f);
                sA[(row0 + hs * 8 + i) * FEAT_STRIDE + nt * 16 + nl] = (_Float16)r;
            }
        }
    }

    // ----- layer 3: sA -> sigmoid(H2 @ W3^T + b3), N=8 valid cols of one tile -----
    {
        const v16h a0 = loadA(sA, 0);
        const v16h a1 = loadA(sA, 1);
        const float bv = (nl < 8) ? b3[nl] : 0.f;
        v8f c;
        #pragma unroll
        for (int i = 0; i < 8; ++i) c[i] = bv;
        c = wmma16x16x32(a0, loadB(sW3, 0, 0), c);
        c = wmma16x16x32(a1, loadB(sW3, 0, 1), c);
        if (nl < 8) {
            #pragma unroll
            for (int i = 0; i < 8; ++i) {
                const long pt =
                    (long)blockIdx.x * PTS_PER_BLOCK + row0 + hs * 8 + i;
                if (pt < nPoints)
                    out[pt * 8 + nl] = 1.f / (1.f + __expf(-c[i]));
            }
        }
    }
}

extern "C" void kernel_launch(void* const* d_in, const int* in_sizes, int n_in,
                              void* d_out, int out_size, void* d_ws, size_t ws_size,
                              hipStream_t stream) {
    const float* ipos   = (const float*)d_in[0];
    const float* tables = (const float*)d_in[1];
    const float* W1     = (const float*)d_in[2];
    const float* b1     = (const float*)d_in[3];
    const float* W2     = (const float*)d_in[4];
    const float* b2     = (const float*)d_in[5];
    const float* W3     = (const float*)d_in[6];
    const float* b3     = (const float*)d_in[7];
    float*       out    = (float*)d_out;

    const int nPoints = in_sizes[0] / 3;
    const int blocks  = (nPoints + PTS_PER_BLOCK - 1) / PTS_PER_BLOCK;

    hipLaunchKernelGGL(hashgrid_mlp_fused, dim3(blocks), dim3(256), 0, stream,
                       ipos, tables, W1, b1, W2, b2, W3, b3, out, nPoints);
}
